// PNANodeModel_22634477649931
// MI455X (gfx1250) — matrix operand order repaired and checked by reading, not compiled, not run on hardware
//
#include <hip/hip_runtime.h>
#include <math.h>

typedef __attribute__((ext_vector_type(16))) _Float16 v16h;
typedef __attribute__((ext_vector_type(8)))  float    v8f;

#define F_DIM 64
#define E_TOT 524288
#define N_MAX 32768
#define B_GR  64
#define BIGF  3.0e38f
// log(17)
#define AVG_LOG_F 2.8332133440562162f

// ---------------- float atomic min/max via int punning (correct total order) --
__device__ __forceinline__ void atomicMaxF(float* a, float v) {
    if (v >= 0.0f) atomicMax((int*)a, __float_as_int(v));
    else           atomicMin((unsigned int*)a, __float_as_uint(v));
}
__device__ __forceinline__ void atomicMinF(float* a, float v) {
    if (v >= 0.0f) atomicMin((int*)a, __float_as_int(v));
    else           atomicMax((unsigned int*)a, __float_as_uint(v));
}

// ---------------- setup / init kernels ---------------------------------------
__global__ void copyf_k(const float* __restrict__ a, float* __restrict__ b, int n) {
    int i = blockIdx.x * blockDim.x + threadIdx.x;
    if (i < n) b[i] = a[i];
}
__global__ void edgesetup_k(const int* __restrict__ ei, int* __restrict__ src,
                            int* __restrict__ dst, int* __restrict__ valid, int E) {
    int e = blockIdx.x * blockDim.x + threadIdx.x;
    if (e < E) { src[e] = ei[e]; dst[e] = ei[E + e]; valid[e] = 1; }
}
__global__ void initacc_k(float* S, float* SQ, float* MN, float* MX, float* CNT, int N) {
    int i = blockIdx.x * blockDim.x + threadIdx.x;
    if (i < N * 64) { S[i] = 0.0f; SQ[i] = 0.0f; MN[i] = BIGF; MX[i] = -BIGF; }
    if (i < N) CNT[i] = 0.0f;
}

// ---------------- weight fusion: Wf = Wpost @ Wlin, bf = bpost@Wlin + blin ---
__global__ void fuse_w_k(const float* __restrict__ Wpost, const float* __restrict__ bpost,
                         const float* __restrict__ Wlin, const float* __restrict__ blin,
                         float* __restrict__ Wf, float* __restrict__ bf) {
    int idx = blockIdx.x * blockDim.x + threadIdx.x;
    if (idx < 1024 * 64) {
        int i = idx >> 6, n = idx & 63;
        float a = 0.0f;
        #pragma unroll 8
        for (int j = 0; j < 64; ++j) a += Wpost[i * 64 + j] * Wlin[j * 64 + n];
        Wf[idx] = a;
    }
    if (idx < 64) {
        float a = blin[idx];
        for (int j = 0; j < 64; ++j) a += bpost[j] * Wlin[j * 64 + idx];
        bf[idx] = a;
    }
}

// ---------------- pack W[K,64] (row-major) into WMMA B fragments (f16) -------
// out[((kb*Ntiles+nb)*32 + lane)*16 + i] = W[kb*32 + (i%8) + 8*(lane/16) + 16*(i/8)][nb*16 + lane%16]
__global__ void pack_b_k(const float* __restrict__ W, int Ktiles, int Ntiles,
                         _Float16* __restrict__ out) {
    int idx = blockIdx.x * blockDim.x + threadIdx.x;
    int total = Ktiles * Ntiles * 32 * 16;
    if (idx >= total) return;
    int i    = idx & 15;
    int lane = (idx >> 4) & 31;
    int t    = idx >> 9;            // kb*Ntiles + nb
    int nb   = t % Ntiles;
    int kb   = t / Ntiles;
    int K = kb * 32 + (i & 7) + 8 * (lane >> 4) + 16 * (i >> 3);
    int N = nb * 16 + (lane & 15);
    out[idx] = (_Float16)W[K * (Ntiles * 16) + N];
}

// ---------------- edge message GEMM (WMMA) + segment aggregation ------------
// Per wave: 16 edges x 64 outputs, K=192 = [x[dst](64) | x[src](64) | edge_enc(64)]
__global__ __launch_bounds__(128) void edge_msg_k(
    const float* __restrict__ x, const int* __restrict__ src, const int* __restrict__ dst,
    const int* __restrict__ valid, const float* __restrict__ eattr,
    const float* __restrict__ We, const float* __restrict__ be,
    const _Float16* __restrict__ Bpack, const float* __restrict__ bpre,
    float* __restrict__ S, float* __restrict__ SQ, float* __restrict__ MN,
    float* __restrict__ MX, float* __restrict__ CNT, int nTiles) {
    __shared__ _Float16 Ash[4][16 * 192];
    int wave = threadIdx.x >> 5;
    int lane = threadIdx.x & 31;
    int tile = blockIdx.x * 4 + wave;
    if (tile >= nTiles) return;
    int e0 = tile * 16;
    _Float16* A = Ash[wave];
    int r = lane & 15, hi = lane >> 4;
    {   // stage 16 rows of [xd | xs | enc] as f16 into LDS
        int e = e0 + r;
        int d = dst[e], sN = src[e];
        const float* xd = x + (size_t)d * 64;
        const float* xs = x + (size_t)sN * 64;
        float ea[8];
        #pragma unroll
        for (int j = 0; j < 8; ++j) ea[j] = eattr[(size_t)e * 8 + j];
        int f0 = hi * 32;
        for (int f = f0; f < f0 + 32; ++f) {
            A[r * 192 + f]      = (_Float16)xd[f];
            A[r * 192 + 64 + f] = (_Float16)xs[f];
            float enc = be[f];
            #pragma unroll
            for (int j = 0; j < 8; ++j) enc += ea[j] * We[j * 64 + f];
            A[r * 192 + 128 + f] = (_Float16)enc;
        }
    }
    asm volatile("s_wait_dscnt 0x0" ::: "memory");
    v8f acc[4] = {};
    #pragma unroll
    for (int kb = 0; kb < 6; ++kb) {
        union { v16h v; unsigned u[8]; } a;
        int m = lane & 15;
        #pragma unroll
        for (int j = 0; j < 8; ++j) {
            int Kl = 2 * (j & 3) + 8 * hi + 16 * (j >> 2);
            a.u[j] = *(const unsigned*)&A[m * 192 + kb * 32 + Kl];
        }
        #pragma unroll
        for (int nb = 0; nb < 4; ++nb) {
            v16h b = *(const v16h*)(Bpack + ((size_t)(kb * 4 + nb) * 32 + lane) * 16);
            acc[nb] = __builtin_amdgcn_wmma_f32_16x16x32_f16(
                false, a.v, false, b, (short)0, acc[nb], false, false);
        }
    }
    // scatter-aggregate: lane owns rows m = v + 8*hi, column n = nb*16 + lane%16
    int col = lane & 15;
    #pragma unroll
    for (int v = 0; v < 8; ++v) {
        int e = e0 + v + 8 * hi;
        if (!valid[e]) continue;
        int d = dst[e];
        #pragma unroll
        for (int nb = 0; nb < 4; ++nb) {
            int n = nb * 16 + col;
            float val = acc[nb][v] + bpre[n];
            float* p = S + (size_t)d * 64 + n;
            atomicAdd(p, val);
            atomicAdd(SQ + (size_t)d * 64 + n, val * val);
            atomicMaxF(MX + (size_t)d * 64 + n, val);
            atomicMinF(MN + (size_t)d * 64 + n, val);
        }
        if (col == 0) atomicAdd(CNT + d, 1.0f);
    }
}

// ---------------- node post GEMM (WMMA): [16,1024] @ Wfused[1024,64] --------
// row = [x(64) | mean,min,max,std,sum (320) | same*amp (320) | same/amp (320)]
__global__ __launch_bounds__(64) void node_post_k(
    const float* __restrict__ x, const float* __restrict__ S, const float* __restrict__ SQ,
    const float* __restrict__ MN, const float* __restrict__ MX, const float* __restrict__ CNT,
    const _Float16* __restrict__ Bpack, const float* __restrict__ bf,
    float* __restrict__ H, int nTiles) {
    __shared__ float Rsh[2][16 * 384];
    __shared__ float Amp[2][16];
    int wave = threadIdx.x >> 5, lane = threadIdx.x & 31;
    int tile = blockIdx.x * 2 + wave;
    if (tile >= nTiles) return;
    float* R = Rsh[wave];
    int r = lane & 15, hi = lane >> 4;
    int node = tile * 16 + r;
    float cnt  = CNT[node];
    float safe = fmaxf(cnt, 1.0f);
    float inv  = 1.0f / safe;
    bool  has  = cnt > 0.0f;
    if (hi == 0) Amp[wave][r] = logf(safe + 1.0f) * (1.0f / AVG_LOG_F);
    int f0 = hi * 32;
    for (int f = f0; f < f0 + 32; ++f) {
        size_t o = (size_t)node * 64 + f;
        float s    = S[o];
        float mean = s * inv;
        float var  = fmaxf(SQ[o] * inv - mean * mean, 0.0f);
        R[r * 384 + f]       = x[o];
        R[r * 384 + 64 + f]  = mean;
        R[r * 384 + 128 + f] = has ? MN[o] : 0.0f;
        R[r * 384 + 192 + f] = has ? MX[o] : 0.0f;
        R[r * 384 + 256 + f] = sqrtf(var + 1e-5f);
        R[r * 384 + 320 + f] = s;
    }
    asm volatile("s_wait_dscnt 0x0" ::: "memory");
    float amp  = Amp[wave][r];
    float ampi = 1.0f / amp;
    v8f acc[4] = {};
    for (int kb = 0; kb < 32; ++kb) {
        int base = kb * 32;                 // region boundaries at 384 & 704 (mult of 32)
        float scale = 1.0f; int ofs = 0;
        if (base >= 704)      { scale = ampi; ofs = -640; }
        else if (base >= 384) { scale = amp;  ofs = -320; }
        union { v16h v; _Float16 h[16]; } a;
        #pragma unroll
        for (int i = 0; i < 16; ++i) {
            int K = base + (i & 7) + 8 * hi + 16 * (i >> 3);
            a.h[i] = (_Float16)(R[r * 384 + K + ofs] * scale);
        }
        #pragma unroll
        for (int nb = 0; nb < 4; ++nb) {
            v16h b = *(const v16h*)(Bpack + ((size_t)(kb * 4 + nb) * 32 + lane) * 16);
            acc[nb] = __builtin_amdgcn_wmma_f32_16x16x32_f16(
                false, a.v, false, b, (short)0, acc[nb], false, false);
        }
    }
    #pragma unroll
    for (int nb = 0; nb < 4; ++nb) {
        int n = nb * 16 + (lane & 15);
        float bv = bf[n];
        #pragma unroll
        for (int v = 0; v < 8; ++v) {
            int m = v + 8 * hi;
            H[(size_t)(tile * 16 + m) * 64 + n] = acc[nb][v] + bv;
        }
    }
}

// ---------------- scoring -----------------------------------------------------
__global__ void wnorm_k(const float* __restrict__ w, float* __restrict__ nrm) {
    if (blockIdx.x == 0 && threadIdx.x == 0) {
        float s = 0.0f;
        for (int i = 0; i < 64; ++i) s += w[i] * w[i];
        *nrm = sqrtf(s);
    }
}
__global__ void score_k(const float* __restrict__ H, const float* __restrict__ w,
                        const float* __restrict__ nrm, float* __restrict__ score, int N) {
    int n = blockIdx.x * blockDim.x + threadIdx.x;
    if (n >= N) return;
    float s = 0.0f;
    #pragma unroll 8
    for (int i = 0; i < 64; ++i) s += H[(size_t)n * 64 + i] * w[i];
    s /= *nrm;
    score[n] = 1.0f / (1.0f + expf(-s));
}

// ---------------- per-graph top-k via LDS bitonic sort ------------------------
// order: score descending, ties -> lower index first (matches jax.lax.top_k)
__global__ __launch_bounds__(256) void topk_k(const float* __restrict__ score,
                                              int n_per, int k, int* __restrict__ perm) {
    __shared__ float ss[512];
    __shared__ int   si[512];
    int b = blockIdx.x;
    for (int i = threadIdx.x; i < n_per; i += 256) { ss[i] = score[b * n_per + i]; si[i] = i; }
    __syncthreads();
    for (int kk = 2; kk <= n_per; kk <<= 1) {
        for (int j = kk >> 1; j > 0; j >>= 1) {
            for (int i = threadIdx.x; i < n_per; i += 256) {
                int ixj = i ^ j;
                if (ixj > i) {
                    float s1 = ss[i], s2 = ss[ixj];
                    int   i1 = si[i], i2 = si[ixj];
                    bool after = (s1 < s2) || (s1 == s2 && i1 > i2); // i ranks after ixj (desc)
                    bool desc  = ((i & kk) == 0);
                    if (desc ? after : !after) {
                        ss[i] = s2; ss[ixj] = s1; si[i] = i2; si[ixj] = i1;
                    }
                }
            }
            __syncthreads();
        }
    }
    for (int i = threadIdx.x; i < k; i += 256) perm[b * k + i] = b * n_per + si[i];
}

// ---------------- pooling bookkeeping ----------------------------------------
__global__ void mapinit_k(int* __restrict__ mapping, int N) {
    int i = blockIdx.x * blockDim.x + threadIdx.x;
    if (i < N) mapping[i] = -1;
}
__global__ void gather_px_k(const float* __restrict__ H, const float* __restrict__ score,
                            const int* __restrict__ perm, float* __restrict__ px,
                            int* __restrict__ mapping, int nkeep) {
    int i = blockIdx.x * blockDim.x + threadIdx.x;
    if (i >= nkeep) return;
    int p = perm[i];
    mapping[p] = i;
    float sc = score[p];
    for (int f = 0; f < 64; ++f) px[(size_t)i * 64 + f] = H[(size_t)p * 64 + f] * sc;
}
__global__ void relabel_k(int* __restrict__ src, int* __restrict__ dst,
                          int* __restrict__ valid, const int* __restrict__ mapping, int E) {
    int e = blockIdx.x * blockDim.x + threadIdx.x;
    if (e >= E) return;
    int v = valid[e];
    int ns = mapping[src[e]], nd = mapping[dst[e]];
    v = (v && ns >= 0 && nd >= 0) ? 1 : 0;
    valid[e] = v;
    src[e] = v ? ns : 0;
    dst[e] = v ? nd : 0;
}

// ---------------- batchnorm (training stats, biased var) + relu ---------------
__global__ __launch_bounds__(256) void bnstats_k(const float* __restrict__ px, int nrows,
                                                 float* __restrict__ stats) {
    int f = blockIdx.x;
    float s = 0.0f, sq = 0.0f;
    for (int r = threadIdx.x; r < nrows; r += 256) {
        float v = px[(size_t)r * 64 + f];
        s += v; sq += v * v;
    }
    __shared__ float sh[256], sh2[256];
    sh[threadIdx.x] = s; sh2[threadIdx.x] = sq;
    __syncthreads();
    for (int off = 128; off > 0; off >>= 1) {
        if (threadIdx.x < off) { sh[threadIdx.x] += sh[threadIdx.x + off];
                                 sh2[threadIdx.x] += sh2[threadIdx.x + off]; }
        __syncthreads();
    }
    if (threadIdx.x == 0) {
        float mu = sh[0] / nrows;
        stats[f] = mu;
        stats[64 + f] = sh2[0] / nrows - mu * mu;
    }
}
__global__ void bnapply_k(const float* __restrict__ px, const float* __restrict__ stats,
                          const float* __restrict__ gamma, const float* __restrict__ beta,
                          float* __restrict__ xout, int total) {
    int i = blockIdx.x * blockDim.x + threadIdx.x;
    if (i >= total) return;
    int f = i & 63;
    float mu = stats[f], var = stats[64 + f];
    float v = gamma[f] * (px[i] - mu) * rsqrtf(var + 1e-5f) + beta[f];
    xout[i] = fmaxf(v, 0.0f);
}

// ---------------- per-graph global max/mean pooling ---------------------------
__global__ __launch_bounds__(64) void pool_k(const float* __restrict__ x, int k,
                                             float* __restrict__ out, int ldo) {
    int b = blockIdx.x, f = threadIdx.x;
    float mx = -BIGF, s = 0.0f;
    for (int r = 0; r < k; ++r) {
        float v = x[(size_t)(b * k + r) * 64 + f];
        mx = fmaxf(mx, v); s += v;
    }
    out[(size_t)b * ldo + f]      = mx;
    out[(size_t)b * ldo + 64 + f] = s / (float)k;
}

// =============================================================================
extern "C" void kernel_launch(void* const* d_in, const int* in_sizes, int n_in,
                              void* d_out, int out_size, void* d_ws, size_t ws_size,
                              hipStream_t stream) {
    const float* x_in  = (const float*)d_in[0];
    const int*   eidx  = (const int*)d_in[1];
    const float* eattr = (const float*)d_in[2];
    const float* We    = (const float*)d_in[4];
    const float* be    = (const float*)d_in[5];
    const float* Wpre  = (const float*)d_in[6];
    const float* bpre  = (const float*)d_in[7];
    const float* Wpost = (const float*)d_in[8];
    const float* bpost = (const float*)d_in[9];
    const float* Wlin  = (const float*)d_in[10];
    const float* blin  = (const float*)d_in[11];
    const float* wtopk = (const float*)d_in[12];
    const float* gamma = (const float*)d_in[13];
    const float* beta  = (const float*)d_in[14];

    const int E = E_TOT, Nmax = N_MAX, Bb = B_GR;

    size_t off = 0;
    auto alloc = [&](size_t bytes) -> void* {
        void* p = (char*)d_ws + off;
        off += (bytes + 255) & ~(size_t)255;
        return p;
    };
    float* xbuf  = (float*)alloc((size_t)Nmax * 64 * 4);
    float* hbuf  = (float*)alloc((size_t)Nmax * 64 * 4);
    float* Sacc  = (float*)alloc((size_t)Nmax * 64 * 4);
    float* SQacc = (float*)alloc((size_t)Nmax * 64 * 4);
    float* MNacc = (float*)alloc((size_t)Nmax * 64 * 4);
    float* MXacc = (float*)alloc((size_t)Nmax * 64 * 4);
    float* CNT   = (float*)alloc((size_t)Nmax * 4);
    int*   srcW  = (int*)alloc((size_t)E * 4);
    int*   dstW  = (int*)alloc((size_t)E * 4);
    int*   valW  = (int*)alloc((size_t)E * 4);
    float* score = (float*)alloc((size_t)Nmax * 4);
    int*   perm  = (int*)alloc((size_t)(Nmax / 2) * 4);
    int*   mapping = (int*)alloc((size_t)Nmax * 4);
    float* px    = (float*)alloc((size_t)(Nmax / 2) * 64 * 4);
    float* Wf    = (float*)alloc((size_t)1024 * 64 * 4);
    float* bf    = (float*)alloc(64 * 4);
    _Float16* BpackPre  = (_Float16*)alloc((size_t)6 * 4 * 32 * 16 * 2);
    _Float16* BpackPost = (_Float16*)alloc((size_t)32 * 4 * 32 * 16 * 2);
    float* bnst  = (float*)alloc(128 * 4);
    float* wnorm = (float*)alloc(4);

    auto cdiv = [](int a, int b) { return (a + b - 1) / b; };

    // copy x to mutable buffer; materialize edge arrays with validity
    copyf_k<<<cdiv(Nmax * 64, 256), 256, 0, stream>>>(x_in, xbuf, Nmax * 64);
    edgesetup_k<<<cdiv(E, 256), 256, 0, stream>>>(eidx, srcW, dstW, valW, E);

    float* feats = (float*)d_out + (size_t)4096 * 64; // feats region [B, 384]

    for (int l = 0; l < 3; ++l) {
        int n_per = 512 >> l;          // 512, 256, 128
        int Nl    = Bb * n_per;        // 32768, 16384, 8192
        int k     = n_per >> 1;
        int nkeep = Bb * k;
        const float* We_l    = We    + (size_t)l * 8 * 64;
        const float* be_l    = be    + (size_t)l * 64;
        const float* Wpre_l  = Wpre  + (size_t)l * 192 * 64;
        const float* bpre_l  = bpre  + (size_t)l * 64;
        const float* Wpost_l = Wpost + (size_t)l * 1024 * 64;
        const float* bpost_l = bpost + (size_t)l * 64;
        const float* Wlin_l  = Wlin  + (size_t)l * 64 * 64;
        const float* blin_l  = blin  + (size_t)l * 64;
        const float* wtopk_l = wtopk + (size_t)l * 64;
        const float* gamma_l = gamma + (size_t)l * 64;
        const float* beta_l  = beta  + (size_t)l * 64;

        initacc_k<<<cdiv(Nl * 64, 256), 256, 0, stream>>>(Sacc, SQacc, MNacc, MXacc, CNT, Nl);
        fuse_w_k<<<cdiv(1024 * 64, 256), 256, 0, stream>>>(Wpost_l, bpost_l, Wlin_l, blin_l, Wf, bf);
        pack_b_k<<<cdiv(6 * 4 * 512, 256), 256, 0, stream>>>(Wpre_l, 6, 4, BpackPre);
        pack_b_k<<<cdiv(32 * 4 * 512, 256), 256, 0, stream>>>(Wf, 32, 4, BpackPost);

        int eTiles = E / 16; // 32768
        edge_msg_k<<<cdiv(eTiles, 4), 128, 0, stream>>>(
            xbuf, srcW, dstW, valW, eattr, We_l, be_l, BpackPre, bpre_l,
            Sacc, SQacc, MNacc, MXacc, CNT, eTiles);

        int nTiles = Nl / 16;
        node_post_k<<<cdiv(nTiles, 2), 64, 0, stream>>>(
            xbuf, Sacc, SQacc, MNacc, MXacc, CNT, BpackPost, bf, hbuf, nTiles);

        wnorm_k<<<1, 32, 0, stream>>>(wtopk_l, wnorm);
        score_k<<<cdiv(Nl, 256), 256, 0, stream>>>(hbuf, wtopk_l, wnorm, score, Nl);
        topk_k<<<Bb, 256, 0, stream>>>(score, n_per, k, perm);

        mapinit_k<<<cdiv(Nl, 256), 256, 0, stream>>>(mapping, Nl);
        gather_px_k<<<cdiv(nkeep, 256), 256, 0, stream>>>(hbuf, score, perm, px, mapping, nkeep);
        relabel_k<<<cdiv(E, 256), 256, 0, stream>>>(srcW, dstW, valW, mapping, E);

        bnstats_k<<<64, 256, 0, stream>>>(px, nkeep, bnst);
        bnapply_k<<<cdiv(nkeep * 64, 256), 256, 0, stream>>>(px, bnst, gamma_l, beta_l,
                                                             xbuf, nkeep * 64);
        pool_k<<<Bb, 64, 0, stream>>>(xbuf, k, feats + (size_t)l * 128, 384);
    }

    // final x: [B*64, 64] -> d_out[0 : 262144]
    hipMemcpyAsync(d_out, xbuf, (size_t)4096 * 64 * 4, hipMemcpyDeviceToDevice, stream);
}
